// CausalAttention_37701222924400
// MI455X (gfx1250) — matrix operand order repaired and checked by reading, not compiled, not run on hardware
//
#include <hip/hip_runtime.h>

// ---------------- problem constants ----------------
#define B_  8
#define T_  4096
#define D_  768
#define H_  12
#define HD_ 64
#define NT_ 32768   // B_*T_

typedef __bf16 bf16_t;
typedef __attribute__((ext_vector_type(16))) __bf16       v16bf;
typedef __attribute__((ext_vector_type(8)))  float        v8f;
typedef __attribute__((ext_vector_type(4)))  unsigned int u32x4;
typedef __attribute__((ext_vector_type(8)))  int          i32x8;
typedef __attribute__((ext_vector_type(4)))  int          i32x4;

// ---------------- helpers ----------------
__device__ inline bf16_t f2bf(float f) {
  unsigned u = __float_as_uint(f);
  u += 0x7fffu + ((u >> 16) & 1u);           // round-to-nearest-even
  unsigned short h = (unsigned short)(u >> 16);
  union { unsigned short s; bf16_t b; } cv; cv.s = h; return cv.b;
}

// A fragment, 16x32 bf16 (M x K), row-major source, row picked by caller.
// lanes 0-15: K {0..7,16..23}; lanes 16-31: K {8..15,24..31}  (ISA 7.12.2)
__device__ inline v16bf fragA(const bf16_t* rowPtr, int k0, int lane) {
  const bf16_t* p = rowPtr + k0 + ((lane & 16) ? 8 : 0);
  union { u32x4 u[2]; v16bf v; } un;
  un.u[0] = *(const u32x4*)p;
  un.u[1] = *(const u32x4*)(p + 16);
  return un.v;
}

// B fragment, 32x16 bf16 (K x N). rowPtr = start of the K-contiguous column
// owned by this lane (n = lane&15). lanes 0-15: K {0..7,16..23}; lanes 16-31: K {8..15,24..31}.
__device__ inline v16bf fragB(const bf16_t* rowPtr, int k0, int lane) {
  const bf16_t* p = rowPtr + k0 + ((lane & 16) ? 8 : 0);
  union { u32x4 u[2]; v16bf v; } un;
  un.u[0] = *(const u32x4*)p;
  un.u[1] = *(const u32x4*)(p + 16);
  return un.v;
}

__device__ inline v8f wmma_bf16(v16bf a, v16bf b, v8f c) {
  return __builtin_amdgcn_wmma_f32_16x16x32_bf16(false, a, false, b, (short)0, c,
                                                 false, false);
}

__device__ inline float bfly_max16(float v, int lane) {
#pragma unroll
  for (int m = 1; m < 16; m <<= 1) {
    float o = __int_as_float(
        __builtin_amdgcn_ds_bpermute(((lane ^ m) << 2), __float_as_int(v)));
    v = fmaxf(v, o);
  }
  return v;
}
__device__ inline float bfly_add16(float v, int lane) {
#pragma unroll
  for (int m = 1; m < 16; m <<= 1) {
    float o = __int_as_float(
        __builtin_amdgcn_ds_bpermute(((lane ^ m) << 2), __float_as_int(v)));
    v += o;
  }
  return v;
}

// --------- TDM: async DMA of a 64-row x 64-elem bf16 tile into LDS ---------
// D# packing per CDNA5 ISA (group0 128b, group1 256b); 2D tensor, so
// descriptor groups 2/3 are zero/disabled.
__device__ inline void tdm_load_tile64x64(unsigned lds_off, const bf16_t* gptr,
                                          unsigned stride_elems) {
  unsigned long long ga = (unsigned long long)(__SIZE_TYPE__)gptr;
  u32x4 g0;
  g0[0] = 1u;                                    // count=1, user mode
  g0[1] = lds_off;                               // lds_addr (bytes)
  g0[2] = (unsigned)(ga & 0xFFFFFFFFu);          // global_addr[31:0]
  g0[3] = (unsigned)((ga >> 32) & 0x01FFFFFFu)   // global_addr[56:32]
          | (2u << 30);                          // type = 2 ("image")
  i32x8 g1;
  g1[0] = (int)(1u << 16);        // workgroup_mask=0, data_size=1 (2 bytes)
  g1[1] = (int)(64u << 16);       // tensor_dim0 = 64 (low 16 in [31:16])
  g1[2] = (int)(64u << 16);       // dim0 hi = 0 | tensor_dim1 = 64
  g1[3] = (int)(64u << 16);       // dim1 hi = 0 | tile_dim0 = 64
  g1[4] = 64;                     // tile_dim1 = 64, tile_dim2 = 0
  g1[5] = (int)stride_elems;      // tensor_dim0_stride[31:0] (data_size units)
  g1[6] = 0;                      // stride0 hi, stride1 lo
  g1[7] = 0;
  i32x4 z4 = {0, 0, 0, 0};
#if defined(__clang_major__) && __clang_major__ >= 23
  i32x8 z8 = {0, 0, 0, 0, 0, 0, 0, 0};
  __builtin_amdgcn_tensor_load_to_lds(g0, g1, z4, z4, z8, 0);
#else
  __builtin_amdgcn_tensor_load_to_lds(g0, g1, z4, z4, 0);
#endif
}

// ---------------- kernel 0: convert x -> bf16, transpose+convert weights ----
__global__ __launch_bounds__(256) void prep_kernel(
    const float* __restrict__ x,
    const float* __restrict__ Wq, const float* __restrict__ Wk,
    const float* __restrict__ Wv, const float* __restrict__ Wo,
    bf16_t* __restrict__ xb,
    bf16_t* __restrict__ WtQ, bf16_t* __restrict__ WtK,
    bf16_t* __restrict__ WtV, bf16_t* __restrict__ WtO) {
  size_t i = (size_t)blockIdx.x * 256 + threadIdx.x;
  if (i < (size_t)NT_ * D_) xb[i] = f2bf(x[i]);
  if (i < (size_t)D_ * D_) {
    int n = (int)(i / D_), k = (int)(i % D_);
    size_t s = (size_t)k * D_ + n;           // Wt[n][k] = W[k][n]
    WtQ[i] = f2bf(Wq[s]);
    WtK[i] = f2bf(Wk[s]);
    WtV[i] = f2bf(Wv[s]);
    WtO[i] = f2bf(Wo[s]);
  }
}

// ---------------- WMMA GEMM: C(Mx768) = A(Mx768 bf16) * Bt^T ----------------
// MODE 0: Q  -> outF f32 (row-major) + per-block sumsq partial
// MODE 1: K  -> same, second partial slot (caller offsets `part`)
// MODE 2: V  -> outF = v1 (B,H,T,HD) f32, outB = V^T (B,H,HD,T) bf16 * (l1+l2)
// MODE 3: O  -> outF = y (row-major f32)
template <int MODE>
__global__ __launch_bounds__(256) void gemm_kernel(
    const bf16_t* __restrict__ A, const bf16_t* __restrict__ Bt,
    float* __restrict__ outF, bf16_t* __restrict__ outB,
    float* __restrict__ part, const float* __restrict__ pl1,
    const float* __restrict__ pl2) {
  __shared__ float red[256];
  const int tid  = threadIdx.x;
  const int lane = tid & 31;
  const int w    = tid >> 5;                 // 8 waves
  const int m0 = blockIdx.y * 128 + (w & 3) * 32;
  const int n0 = blockIdx.x * 64  + (w >> 2) * 32;

  const bf16_t* arow0 = A  + (size_t)(m0 + (lane & 15)) * D_;
  const bf16_t* arow1 = arow0 + (size_t)16 * D_;
  const bf16_t* brow0 = Bt + (size_t)(n0 + (lane & 15)) * D_;
  const bf16_t* brow1 = brow0 + (size_t)16 * D_;

  const v8f vz = {0.f, 0.f, 0.f, 0.f, 0.f, 0.f, 0.f, 0.f};
  v8f c[2][2] = {{vz, vz}, {vz, vz}};

  for (int k0 = 0; k0 < D_; k0 += 32) {
    v16bf a0 = fragA(arow0, k0, lane);
    v16bf a1 = fragA(arow1, k0, lane);
    v16bf b0 = fragB(brow0, k0, lane);
    v16bf b1 = fragB(brow1, k0, lane);
    c[0][0] = wmma_bf16(a0, b0, c[0][0]);
    c[0][1] = wmma_bf16(a0, b1, c[0][1]);
    c[1][0] = wmma_bf16(a1, b0, c[1][0]);
    c[1][1] = wmma_bf16(a1, b1, c[1][1]);
  }

  float lam = 0.f;
  if (MODE == 2) lam = pl1[0] + pl2[0];      // v = (l1+l2)*v
  float ss = 0.f;
  const int rbase = (lane & 16) ? 8 : 0;     // C layout: row = vgpr + rbase
  const int ncol  = lane & 15;

#pragma unroll
  for (int ti = 0; ti < 2; ++ti)
#pragma unroll
    for (int tj = 0; tj < 2; ++tj)
#pragma unroll
      for (int r = 0; r < 8; ++r) {
        int m = m0 + ti * 16 + rbase + r;
        int n = n0 + tj * 16 + ncol;
        float val = c[ti][tj][r];
        if (MODE == 0 || MODE == 1) {
          outF[(size_t)m * D_ + n] = val;
          ss += val * val;
        } else if (MODE == 2) {
          int b = m >> 12, t = m & 4095, hh = n >> 6, hd = n & 63;
          outF[(((size_t)(b * H_ + hh) * T_) + t) * HD_ + hd] = val;   // v1
          outB[(((size_t)(b * H_ + hh)) * HD_ + hd) * T_ + t] = f2bf(val * lam);
        } else {
          outF[(size_t)m * D_ + n] = val;     // y
        }
      }

  if (MODE == 0 || MODE == 1) {              // deterministic block tree reduce
    red[tid] = ss;
    __syncthreads();
    for (int s = 128; s > 0; s >>= 1) {
      if (tid < s) red[tid] += red[tid + s];
      __syncthreads();
    }
    if (tid == 0) part[blockIdx.y * gridDim.x + blockIdx.x] = red[0];
  }
}

// ---------------- deterministic fixed-order partial reduction ---------------
__global__ __launch_bounds__(256) void reduce_kernel(const float* __restrict__ part,
                                                     float* __restrict__ sumsq) {
  __shared__ float red[256];
  const float* p = part + blockIdx.x * 3072;
  float s = 0.f;
  for (int i = threadIdx.x; i < 3072; i += 256) s += p[i];
  red[threadIdx.x] = s;
  __syncthreads();
  for (int k = 128; k > 0; k >>= 1) {
    if (threadIdx.x < k) red[threadIdx.x] += red[threadIdx.x + k];
    __syncthreads();
  }
  if (threadIdx.x == 0) sumsq[blockIdx.x] = red[0];
}

// ---------------- RoPE + global RMS scale -> bf16 (RoPE preserves RMS) ------
__global__ __launch_bounds__(256) void rope_norm_kernel(
    const float* __restrict__ Qf, const float* __restrict__ Kf,
    const float* __restrict__ cosT, const float* __restrict__ sinT,
    const float* __restrict__ sumsq, bf16_t* __restrict__ Qb,
    bf16_t* __restrict__ Kb) {
  size_t idx = (size_t)blockIdx.x * 256 + threadIdx.x;
  if (idx >= (size_t)NT_ * (D_ / 2)) return;
  const int g = blockIdx.y;                  // 0 = q, 1 = k
  int row = (int)(idx / (D_ / 2));
  int p   = (int)(idx % (D_ / 2));
  int h = p >> 5, d = p & 31;
  int t = row & (T_ - 1);
  const float* src = g ? Kf : Qf;
  bf16_t*      dst = g ? Kb : Qb;
  float rms = sqrtf(sumsq[g] * (1.0f / ((float)NT_ * (float)D_)));
  float inv = 1.0f / (rms + 1e-8f);
  if (!g) inv *= 0.125f;                     // fold 1/sqrt(HD) into q
  size_t base = (size_t)row * D_ + h * HD_ + d;
  float x1 = src[base], x2 = src[base + 32];
  float c = cosT[t * 32 + d], s = sinT[t * 32 + d];
  dst[base]      = f2bf((x1 * c + x2 * s) * inv);
  dst[base + 32] = f2bf((x2 * c - x1 * s) * inv);
}

// ---------------- flash attention: 4 waves x 16 queries, 64-key tiles -------
// K/V tiles are DMA'd into double-buffered LDS by the Tensor Data Mover
// (wave 0 issues descriptors, TENSORcnt + block barrier for hand-off), so the
// hot loop issues only ds_load_b128 + v_wmma + softmax VALU.
__global__ __launch_bounds__(128) void attn_kernel(
    const bf16_t* __restrict__ Qb, const bf16_t* __restrict__ Kb,
    const bf16_t* __restrict__ Vt, bf16_t* __restrict__ Ob) {
  __shared__ bf16_t ldsK[2][64 * 64];        // 2 x 8 KB (keys x hd)
  __shared__ bf16_t ldsV[2][64 * 64];        // 2 x 8 KB (hd x keys)
  __shared__ bf16_t ldsP[4][16 * 64];        // per-wave P staging (8 KB)
  const int tid  = threadIdx.x;
  const int w    = tid >> 5;
  const int lane = tid & 31;
  const int bh = blockIdx.y;
  const int b = bh / H_, h = bh % H_;

  const bf16_t* Qbase = Qb + ((size_t)b * T_) * D_ + h * HD_;
  const bf16_t* Kg    = Kb + ((size_t)b * T_) * D_ + h * HD_;  // rows=keys
  const bf16_t* Vg    = Vt + ((size_t)bh * HD_) * T_;          // rows=hd

  const int q0   = blockIdx.x * 64 + w * 16;
  const int arow = q0 + (lane & 15);
  const v16bf aq0 = fragA(Qbase + (size_t)arow * D_, 0,  lane);
  const v16bf aq1 = fragA(Qbase + (size_t)arow * D_, 32, lane);

  const v8f vz = {0.f, 0.f, 0.f, 0.f, 0.f, 0.f, 0.f, 0.f};
  v8f o[4] = {vz, vz, vz, vz};
  float mrow[8], lrow[8];
#pragma unroll
  for (int r = 0; r < 8; ++r) { mrow[r] = -1e30f; lrow[r] = 0.f; }

  const int rbase = (lane & 16) ? 8 : 0;
  const int ncol  = lane & 15;
  bf16_t* lp = &ldsP[w][0];
  const int nT = T_ / 64;                    // 64 key tiles

  auto issue_tile = [&](int jt, int buf) {
    tdm_load_tile64x64((unsigned)(__SIZE_TYPE__)(void*)&ldsK[buf][0],
                       Kg + (size_t)(jt * 64) * D_, D_);
    tdm_load_tile64x64((unsigned)(__SIZE_TYPE__)(void*)&ldsV[buf][0],
                       Vg + jt * 64, T_);
  };

  if (w == 0) { issue_tile(0, 0); issue_tile(1, 1); }   // prime double buffer

  for (int j = 0; j < nT; ++j) {
    if (w == 0) {
      if (j + 1 < nT) __builtin_amdgcn_s_wait_tensorcnt(2);  // tile j done
      else            __builtin_amdgcn_s_wait_tensorcnt(0);
    }
    __syncthreads();                         // tile j visible to all waves
    const bf16_t* Kt  = &ldsK[j & 1][0];
    const bf16_t* Vtl = &ldsV[j & 1][0];

    // ---- S = Q K^T (scale already folded into Q) ----
    v8f s[4];
#pragma unroll
    for (int nc = 0; nc < 4; ++nc) {
      const bf16_t* krow = Kt + (nc * 16 + ncol) * 64;
      v8f acc = wmma_bf16(aq0, fragB(krow, 0, lane), vz);
      s[nc]   = wmma_bf16(aq1, fragB(krow, 32, lane), acc);
    }
    // ---- online softmax ----
    float mnew[8], sc[8];
#pragma unroll
    for (int r = 0; r < 8; ++r) {
      float cand = fmaxf(fmaxf(s[0][r], s[1][r]), fmaxf(s[2][r], s[3][r]));
      cand = bfly_max16(cand, lane);
      mnew[r] = fmaxf(mrow[r], cand);
      sc[r]   = __expf(mrow[r] - mnew[r]);
      mrow[r] = mnew[r];
    }
#pragma unroll
    for (int r = 0; r < 8; ++r) {
      float p0 = __expf(s[0][r] - mnew[r]);
      float p1 = __expf(s[1][r] - mnew[r]);
      float p2 = __expf(s[2][r] - mnew[r]);
      float p3 = __expf(s[3][r] - mnew[r]);
      s[0][r] = p0; s[1][r] = p1; s[2][r] = p2; s[3][r] = p3;
      float ps = bfly_add16(p0 + p1 + p2 + p3, lane);
      lrow[r] = lrow[r] * sc[r] + ps;
    }
#pragma unroll
    for (int nc = 0; nc < 4; ++nc)
#pragma unroll
      for (int r = 0; r < 8; ++r) o[nc][r] *= sc[r];
    // ---- P (C-layout f32) -> bf16 A-fragment via per-wave LDS ----
#pragma unroll
    for (int t4 = 0; t4 < 4; ++t4)
#pragma unroll
      for (int r = 0; r < 8; ++r)
        lp[(rbase + r) * 64 + t4 * 16 + ncol] = f2bf(s[t4][r]);
    asm volatile("s_wait_dscnt 0x0" ::: "memory");
    v16bf pa0 = fragA(lp + (lane & 15) * 64, 0,  lane);
    v16bf pa1 = fragA(lp + (lane & 15) * 64, 32, lane);
    // ---- O += P V  (V^T tile rows are key-contiguous in LDS) ----
#pragma unroll
    for (int nc = 0; nc < 4; ++nc) {
      const bf16_t* vrow = Vtl + (nc * 16 + ncol) * 64;
      o[nc] = wmma_bf16(pa0, fragB(vrow, 0, lane), o[nc]);
      o[nc] = wmma_bf16(pa1, fragB(vrow, 32, lane), o[nc]);
    }
    __syncthreads();                         // all waves done with buffer j&1
    if (w == 0 && j + 2 < nT) issue_tile(j + 2, j & 1);
  }

  float linv[8];
#pragma unroll
  for (int r = 0; r < 8; ++r) linv[r] = 1.0f / lrow[r];
#pragma unroll
  for (int nc = 0; nc < 4; ++nc)
#pragma unroll
    for (int r = 0; r < 8; ++r) {
      int trow = q0 + rbase + r;
      Ob[((size_t)b * T_ + trow) * D_ + h * HD_ + nc * 16 + ncol] =
          f2bf(o[nc][r] * linv[r]);
    }
}

// ---------------- host launcher ----------------
extern "C" void kernel_launch(void* const* d_in, const int* in_sizes, int n_in,
                              void* d_out, int out_size, void* d_ws,
                              size_t ws_size, hipStream_t stream) {
  (void)in_sizes; (void)n_in; (void)out_size; (void)ws_size;
  const float* x    = (const float*)d_in[0];
  const float* cosT = (const float*)d_in[1];
  const float* sinT = (const float*)d_in[2];
  const float* Wq   = (const float*)d_in[3];
  const float* Wk   = (const float*)d_in[4];
  const float* Wv   = (const float*)d_in[5];
  const float* Wo   = (const float*)d_in[6];
  const float* l1   = (const float*)d_in[7];
  const float* l2   = (const float*)d_in[8];

  float* y     = (float*)d_out;                     // (B,T,D)
  float* v1out = y + (size_t)NT_ * D_;              // (B,H,T,HD)

  char* w = (char*)d_ws;
  size_t off = 0;
  auto alloc = [&](size_t bytes) {
    size_t cur = off;
    off += (bytes + 255) & ~(size_t)255;
    return cur;
  };
  float*  Qf  = (float*)(w + alloc((size_t)NT_ * D_ * 4));
  float*  Kf  = (float*)(w + alloc((size_t)NT_ * D_ * 4));
  bf16_t* xb  = (bf16_t*)(w + alloc((size_t)NT_ * D_ * 2));
  bf16_t* Qb  = (bf16_t*)(w + alloc((size_t)NT_ * D_ * 2));
  bf16_t* Kb  = (bf16_t*)(w + alloc((size_t)NT_ * D_ * 2));
  bf16_t* Vt  = (bf16_t*)(w + alloc((size_t)NT_ * D_ * 2));
  bf16_t* WtQ = (bf16_t*)(w + alloc((size_t)D_ * D_ * 2));
  bf16_t* WtK = (bf16_t*)(w + alloc((size_t)D_ * D_ * 2));
  bf16_t* WtV = (bf16_t*)(w + alloc((size_t)D_ * D_ * 2));
  bf16_t* WtO = (bf16_t*)(w + alloc((size_t)D_ * D_ * 2));
  float*  part  = (float*)(w + alloc(2 * 3072 * 4));
  float*  sumsq = (float*)(w + alloc(256));
  bf16_t* Ob = (bf16_t*)Qf;   // reuse Qf32 region: dead after rope_norm

  const int prep_blocks = (int)(((size_t)NT_ * D_ + 255) / 256);
  prep_kernel<<<prep_blocks, 256, 0, stream>>>(x, Wq, Wk, Wv, Wo, xb, WtQ, WtK,
                                               WtV, WtO);

  dim3 gg(D_ / 64, NT_ / 128);                      // 12 x 256
  gemm_kernel<0><<<gg, 256, 0, stream>>>(xb, WtQ, Qf, nullptr, part, l1, l2);
  gemm_kernel<1><<<gg, 256, 0, stream>>>(xb, WtK, Kf, nullptr, part + 3072, l1, l2);
  gemm_kernel<2><<<gg, 256, 0, stream>>>(xb, WtV, v1out, Vt, nullptr, l1, l2);

  reduce_kernel<<<2, 256, 0, stream>>>(part, sumsq);

  dim3 gr((unsigned)(((size_t)NT_ * (D_ / 2) + 255) / 256), 2);
  rope_norm_kernel<<<gr, 256, 0, stream>>>(Qf, Kf, cosT, sinT, sumsq, Qb, Kb);

  dim3 ga(T_ / 64, B_ * H_);                        // 64 x 96
  attn_kernel<<<ga, 128, 0, stream>>>(Qb, Kb, Vt, Ob);

  gemm_kernel<3><<<gg, 256, 0, stream>>>(Ob, WtO, y, nullptr, nullptr, l1, l2);
}
